// EvoPFBlock_23081154248869
// MI455X (gfx1250) — compile-verified, matrix operand order checked by reading, not run on hardware
//
#include <hip/hip_runtime.h>
#include <hip/hip_bf16.h>
#include <math.h>

typedef __attribute__((ext_vector_type(16))) _Float16 v16h;
typedef __attribute__((ext_vector_type(8)))  float    v8f;

#define DM   256
#define NSEQ 128
#define LSEQ 512
#define NH   8
#define DH   32
#define ROWS (LSEQ*NSEQ)                 // 65536
#define QSCALE 0.17677669529663688f      // 1/sqrt(32)

// CDNA5 16-bit 16x32 WMMA fragment mapping (ISA 7.12.2).
// Element j (0..15) of the per-lane v16h holds K index:
__device__ __forceinline__ int frag_k(int lane, int j) {
  int kb = (lane >> 4) << 3;             // 0 or 8
  int p = j >> 1, h = j & 1;
  return (p < 4) ? (kb + 2 * p + h) : (16 + kb + 2 * (p - 4) + h);
}
// Inverse: (row-or-col mn in 0..15, k in 0..31) -> offset inside a 512-elt tile
// stored in fragment order (tile + lane*16 + j).
__device__ __forceinline__ int swz_idx(int mn, int k) {
  int lane = mn + (((k >> 3) & 1) << 4);
  int j = ((k >> 4) << 3) | (k & 7);
  return lane * 16 + j;
}
// One aligned 32B load per lane = the whole fragment.
__device__ __forceinline__ v16h load_frag(const _Float16* tile, int lane) {
  return *(const v16h*)(tile + lane * 16);
}

__device__ __forceinline__ v8f wmma_f16(v16h a, v16h b, v8f c) {
  return __builtin_amdgcn_wmma_f32_16x16x32_f16(false, a, false, b,
                                                (short)0, c, false, false);
}

// ---------------------------------------------------------------- weights f32 -> f16 (swizzled B tiles)
// W (K=256 x N=256) -> tiles [nt(16)][kt(8)] of 512, fragment order.
__global__ void k_wcvt(const float* __restrict__ wq, const float* __restrict__ wk,
                       const float* __restrict__ wv, const float* __restrict__ wg,
                       const float* __restrict__ wo, _Float16* __restrict__ dst) {
  int id = blockIdx.x * 256 + threadIdx.x;   // 5*65536 total
  int which = id >> 16, off = id & 65535;
  int k = off >> 8, n = off & 255;           // row-major W[k][n]
  float v;
  if      (which == 0) v = wq[off];
  else if (which == 1) v = wk[off];
  else if (which == 2) v = wv[off];
  else if (which == 3) v = wg[off];
  else                 v = wo[off];
  size_t o = (size_t)which * DM * DM +
             ((size_t)(n >> 4) * 8 + (k >> 5)) * 512 + swz_idx(n & 15, k & 31);
  dst[o] = (_Float16)v;
}

// ---------------------------------------------------------------- LN over channels + transpose
// msa (d_m, n, L) -> x16 as swizzled A tiles: [rowtile(4096)][kt(8)] of 512.
__global__ void k_ln_msa(const float* __restrict__ msa,
                         const float* __restrict__ g, const float* __restrict__ b,
                         _Float16* __restrict__ x16) {
  __shared__ float tile[DM * 32];          // 32 KB: 256 channels x 32 l-values
  __shared__ float s_ps[8 * 32], s_pq[8 * 32];
  __shared__ float s_mean[32], s_inv[32];
  int i   = blockIdx.x;                    // sequence index 0..127
  int l0  = blockIdx.y * 32;               // l tile
  int tid = threadIdx.x, lane = tid & 31, wave = tid >> 5;

  for (int c0 = 0; c0 < DM; c0 += 8) {     // coalesced 128B rows along L
    int c = c0 + wave;
    tile[c * 32 + lane] = msa[(size_t)c * (NSEQ * LSEQ) + (size_t)i * LSEQ + l0 + lane];
  }
  __syncthreads();

  { float s = 0.f, q = 0.f;
    for (int c = wave * 32; c < wave * 32 + 32; ++c) {
      float v = tile[c * 32 + lane]; s += v; q += v * v;
    }
    s_ps[wave * 32 + lane] = s; s_pq[wave * 32 + lane] = q;
  }
  __syncthreads();
  if (tid < 32) {
    float s = 0.f, q = 0.f;
    for (int seg = 0; seg < 8; ++seg) { s += s_ps[seg * 32 + tid]; q += s_pq[seg * 32 + tid]; }
    float m = s * (1.0f / DM);
    float v = q * (1.0f / DM) - m * m;
    s_mean[tid] = m; s_inv[tid] = rsqrtf(v + 1e-5f);
  }
  __syncthreads();

  for (int col = wave * 4; col < wave * 4 + 4; ++col) {
    float m = s_mean[col], inv = s_inv[col];
    int row = (l0 + col) * NSEQ + i;                       // x row (l*n + i)
    size_t rbase = ((size_t)(row >> 4) * 8) * 512;
    int msub = row & 15;
    for (int k = 0; k < 8; ++k) {
      int c = k * 32 + lane;
      float v = (tile[c * 32 + col] - m) * inv * g[c] + b[c];
      x16[rbase + (size_t)(c >> 5) * 512 + swz_idx(msub, c & 31)] = (_Float16)v;
    }
  }
}

// ---------------------------------------------------------------- pair bias
__global__ void k_pair_bias(const float* __restrict__ pairs,
                            const float* __restrict__ g, const float* __restrict__ b,
                            const float* __restrict__ Wb,
                            float* __restrict__ bias) {
  int tid = threadIdx.x, lane = tid & 31, wave = tid >> 5;
  int pidx = blockIdx.x * 8 + wave;        // pidx = kk*128 + qq
  int kk = pidx >> 7, qq = pidx & 127;
  float v[4]; float s = 0.f, q = 0.f;
  for (int j = 0; j < 4; ++j) {
    v[j] = pairs[(size_t)pidx * 128 + lane + 32 * j];
    s += v[j]; q += v[j] * v[j];
  }
  for (int m = 16; m >= 1; m >>= 1) { s += __shfl_xor(s, m, 32); q += __shfl_xor(q, m, 32); }
  float mean = s * (1.f / 128.f);
  float var  = q * (1.f / 128.f) - mean * mean;
  float inv  = rsqrtf(var + 1e-5f);
  float xn[4];
  for (int j = 0; j < 4; ++j) {
    int ch = lane + 32 * j;
    xn[j] = (v[j] - mean) * inv * g[ch] + b[ch];
  }
  for (int h = 0; h < NH; ++h) {
    float p = 0.f;
    for (int j = 0; j < 4; ++j) { int ch = lane + 32 * j; p += xn[j] * Wb[ch * NH + h]; }
    for (int m = 16; m >= 1; m >>= 1) p += __shfl_xor(p, m, 32);
    if (lane == 0) bias[(size_t)h * NSEQ * NSEQ + qq * NSEQ + kk] = p;
  }
}

// ---------------------------------------------------------------- projection GEMM (templated on weight)
// x16 swizzled-A (65536x256) @ W swizzled-B (256x256) -> q/k/v (swizzled) or g (linear).
template <int WHICH>
__global__ void k_proj(const _Float16* __restrict__ x16,
                       const _Float16* __restrict__ W,
                       const float* __restrict__ bg,
                       _Float16* __restrict__ dst) {
  int tid = threadIdx.x, lane = tid & 31, wave = tid >> 5;
  int mt = wave & 1;                       // 2 M-tiles of 16 per block
  int ntbase = (wave >> 1) * 4;            // 4 N-tiles per wave
  int mtile = blockIdx.x * 2 + mt;
  v8f acc[4] = {};
  for (int kb = 0; kb < 8; ++kb) {
    const _Float16* atile = x16 + ((size_t)mtile * 8 + kb) * 512;
    if (kb < 7)
      __builtin_prefetch(atile + 512 + lane * 16, 0, 3);
    v16h a = load_frag(atile, lane);
    #pragma unroll
    for (int t = 0; t < 4; ++t) {
      v16h bf = load_frag(W + ((size_t)(ntbase + t) * 8 + kb) * 512, lane);
      acc[t] = wmma_f16(a, bf, acc[t]);
    }
  }
  #pragma unroll
  for (int t = 0; t < 4; ++t) {
    int ncol = (ntbase + t) * 16 + (lane & 15);
    int h = ncol >> 5, d = ncol & 31;
    #pragma unroll
    for (int r = 0; r < 8; ++r) {
      int mrow = blockIdx.x * 32 + mt * 16 + r + ((lane >> 4) << 3);
      int l = mrow >> 7, ii = mrow & 127;
      size_t lhbase = (size_t)(l * NH + h) * 8 * 512;     // 4096 elems per (l,h)
      float vacc = acc[t][r];
      if (WHICH == 0) {        // q: swizzled-A rows(ii) x d, scaled
        dst[lhbase + (size_t)(ii >> 4) * 512 + swz_idx(ii & 15, d)] =
            (_Float16)(vacc * QSCALE);
      } else if (WHICH == 1) { // k: same tile layout as q
        dst[lhbase + (size_t)(ii >> 4) * 512 + swz_idx(ii & 15, d)] = (_Float16)vacc;
      } else if (WHICH == 2) { // v: swizzled-B (k=key ii, n=d): tile kt*2+nt
        dst[lhbase + (size_t)((ii >> 5) * 2 + (d >> 4)) * 512 +
            swz_idx(d & 15, ii & 31)] = (_Float16)vacc;
      } else {                 // g: linear (l,h,n,d_h), sigmoid
        float sg = 1.f / (1.f + __expf(-(vacc + bg[ncol])));
        dst[((size_t)(l * NH + h) * NSEQ + ii) * DH + d] = (_Float16)sg;
      }
    }
  }
}

// ---------------------------------------------------------------- attention per (l, h)
__global__ void k_attn(const _Float16* __restrict__ q16, const _Float16* __restrict__ k16,
                       const _Float16* __restrict__ v16, const _Float16* __restrict__ g16,
                       const float* __restrict__ bias, _Float16* __restrict__ o16) {
  extern __shared__ char smem[];
  float*    s_sc = (float*)smem;                          // 128x128 f32 = 64 KB
  _Float16* s_w  = (_Float16*)(smem + NSEQ * NSEQ * 4);   // swizzled A tiles, 32 KB
  int l = blockIdx.x, h = blockIdx.y;
  int tid = threadIdx.x, lane = tid & 31, wave = tid >> 5;
  size_t fbase = (size_t)(l * NH + h) * 8 * 512;          // q/k/v fragment base
  int mt = wave;                                          // 8 waves = 8 M-tiles

  // scores = q.k^T + bias ; K = d_h = 32 -> one WMMA per 16x16 tile
  v16h a = load_frag(q16 + fbase + (size_t)mt * 512, lane);
  const float* biash = bias + (size_t)h * NSEQ * NSEQ;
  for (int nt = 0; nt < 8; ++nt) {
    v16h bf = load_frag(k16 + fbase + (size_t)nt * 512, lane);
    v8f c = {};
    c = wmma_f16(a, bf, c);
    #pragma unroll
    for (int r = 0; r < 8; ++r) {
      int row = mt * 16 + r + ((lane >> 4) << 3);
      int col = nt * 16 + (lane & 15);
      s_sc[row * NSEQ + col] = c[r] + biash[row * NSEQ + col];
    }
  }
  __syncthreads();

  // row softmax: f32 scores -> f16 weights written in fragment-swizzled order
  if (tid < NSEQ) {
    float* rowp = s_sc + tid * NSEQ;
    float mx = -1e30f;
    for (int c = 0; c < NSEQ; ++c) mx = fmaxf(mx, rowp[c]);
    float sum = 0.f;
    for (int c = 0; c < NSEQ; ++c) sum += __expf(rowp[c] - mx);
    float inv = 1.f / sum;
    int rt = tid >> 4, msub = tid & 15;
    for (int c = 0; c < NSEQ; ++c)
      s_w[(rt * 4 + (c >> 5)) * 512 + swz_idx(msub, c & 31)] =
          (_Float16)(__expf(rowp[c] - mx) * inv);
  }
  __syncthreads();

  // o = w @ v  (M=128, N=32, K=128), gate by g
  for (int nt = 0; nt < 2; ++nt) {
    v8f acc = {};
    for (int kt = 0; kt < 4; ++kt) {
      v16h aw = load_frag(s_w + (size_t)(mt * 4 + kt) * 512, lane);
      v16h bv = load_frag(v16 + fbase + (size_t)(kt * 2 + nt) * 512, lane);
      acc = wmma_f16(aw, bv, acc);
    }
    #pragma unroll
    for (int r = 0; r < 8; ++r) {
      int msub = r + ((lane >> 4) << 3);
      int row = mt * 16 + msub;
      int d = nt * 16 + (lane & 15);
      float gv = (float)g16[((size_t)(l * NH + h) * NSEQ + row) * DH + d];
      // o16 as swizzled-A tiles over (65536 x 256): rowtile = l*8+mt, ktile = h
      o16[((size_t)(l * 8 + mt) * 8 + h) * 512 + swz_idx(msub, d)] =
          (_Float16)(acc[r] * gv);
    }
  }
}

// ---------------------------------------------------------------- output GEMM + transposed store
__global__ void k_out(const _Float16* __restrict__ o16, const _Float16* __restrict__ Wo16,
                      const float* __restrict__ bo, float* __restrict__ out) {
  int tid = threadIdx.x, lane = tid & 31, wave = tid >> 5;
  int mt = wave & 1;
  int ntbase = (wave >> 1) * 4;
  int mtile = blockIdx.x * 2 + mt;
  v8f acc[4] = {};
  for (int kb = 0; kb < 8; ++kb) {
    const _Float16* atile = o16 + ((size_t)mtile * 8 + kb) * 512;
    if (kb < 7)
      __builtin_prefetch(atile + 512 + lane * 16, 0, 3);
    v16h a = load_frag(atile, lane);
    #pragma unroll
    for (int t = 0; t < 4; ++t) {
      v16h bf = load_frag(Wo16 + ((size_t)(ntbase + t) * 8 + kb) * 512, lane);
      acc[t] = wmma_f16(a, bf, acc[t]);
    }
  }
  #pragma unroll
  for (int t = 0; t < 4; ++t) {
    int c = (ntbase + t) * 16 + (lane & 15);
    float bc = bo[c];
    #pragma unroll
    for (int r = 0; r < 8; ++r) {
      int r2 = blockIdx.x * 32 + mt * 16 + r + ((lane >> 4) << 3);
      int l = r2 >> 7, ii = r2 & 127;
      out[(size_t)c * (NSEQ * LSEQ) + (size_t)ii * LSEQ + l] = acc[t][r] + bc;
    }
  }
}

extern "C" void kernel_launch(void* const* d_in, const int* in_sizes, int n_in,
                              void* d_out, int out_size, void* d_ws, size_t ws_size,
                              hipStream_t stream) {
  (void)in_sizes; (void)n_in; (void)out_size; (void)ws_size;
  const float* msa    = (const float*)d_in[0];
  const float* pairs  = (const float*)d_in[1];
  const float* ln_m_g = (const float*)d_in[2];
  const float* ln_m_b = (const float*)d_in[3];
  const float* ln_p_g = (const float*)d_in[4];
  const float* ln_p_b = (const float*)d_in[5];
  const float* Wq = (const float*)d_in[6];
  const float* Wk = (const float*)d_in[7];
  const float* Wv = (const float*)d_in[8];
  const float* Wg = (const float*)d_in[9];
  const float* bg = (const float*)d_in[10];
  const float* Wb = (const float*)d_in[11];
  const float* Wo = (const float*)d_in[12];
  const float* bo = (const float*)d_in[13];
  float* out = (float*)d_out;

  char* p = (char*)d_ws;
  _Float16* x16 = (_Float16*)p; p += (size_t)ROWS * DM * sizeof(_Float16);     // 32 MB
  _Float16* w16 = (_Float16*)p; p += (size_t)5 * DM * DM * sizeof(_Float16);   // 640 KB
  float* biasbuf = (float*)p;   p += (size_t)NH * NSEQ * NSEQ * sizeof(float); // 512 KB
  _Float16* q16 = (_Float16*)p; p += (size_t)ROWS * DM * sizeof(_Float16);
  _Float16* k16 = (_Float16*)p; p += (size_t)ROWS * DM * sizeof(_Float16);
  _Float16* v16 = (_Float16*)p; p += (size_t)ROWS * DM * sizeof(_Float16);
  _Float16* g16 = (_Float16*)p; p += (size_t)ROWS * DM * sizeof(_Float16);
  _Float16* o16 = (_Float16*)p; p += (size_t)ROWS * DM * sizeof(_Float16);

  k_wcvt<<<dim3(5 * DM * DM / 256), dim3(256), 0, stream>>>(Wq, Wk, Wv, Wg, Wo, w16);
  k_ln_msa<<<dim3(NSEQ, LSEQ / 32), dim3(256), 0, stream>>>(msa, ln_m_g, ln_m_b, x16);
  k_pair_bias<<<dim3(NSEQ * NSEQ / 8), dim3(256), 0, stream>>>(pairs, ln_p_g, ln_p_b, Wb, biasbuf);
  k_proj<0><<<dim3(ROWS / 32), dim3(256), 0, stream>>>(x16, w16 + 0 * (size_t)DM * DM, bg, q16);
  k_proj<1><<<dim3(ROWS / 32), dim3(256), 0, stream>>>(x16, w16 + 1 * (size_t)DM * DM, bg, k16);
  k_proj<2><<<dim3(ROWS / 32), dim3(256), 0, stream>>>(x16, w16 + 2 * (size_t)DM * DM, bg, v16);
  k_proj<3><<<dim3(ROWS / 32), dim3(256), 0, stream>>>(x16, w16 + 3 * (size_t)DM * DM, bg, g16);
  k_attn<<<dim3(LSEQ, NH), dim3(256), NSEQ * NSEQ * 4 + NSEQ * NSEQ * 2, stream>>>(
      q16, k16, v16, g16, biasbuf, o16);
  k_out<<<dim3(ROWS / 32), dim3(256), 0, stream>>>(o16, w16 + 4 * (size_t)DM * DM, bo, out);
}
